// GAttention_3985729651210
// MI455X (gfx1250) — compile-verified, matrix operand order checked by reading, not run on hardware
//
#include <hip/hip_runtime.h>

#define N_DIM 1024
#define IN_DIM 1024
#define H_DIM 64
#define B_DIM 16
#define SROW (B_DIM * IN_DIM) // stride between n-rows of s

typedef __attribute__((ext_vector_type(2))) float v2f;
typedef __attribute__((ext_vector_type(8))) float v8f;
typedef __attribute__((ext_vector_type(8))) __bf16 v8bf;
typedef __attribute__((ext_vector_type(16))) __bf16 v16bf;
typedef int v4i_vs __attribute__((vector_size(16))); // matches builtin param

#define LDA 20 // padded LDS row (floats) for f32 A tiles (64 x 16)
#define LDB 68 // padded LDS row (floats) for f32 B tiles (16 x 64)
#define LDK 40 // padded LDS row (bf16) for 32-wide K tiles (64 x 32)
#define LDT 72 // padded LDS row (bf16) for 64x64 transpose tiles

#if __has_builtin(__builtin_amdgcn_global_load_async_to_lds_b128) && \
    __has_builtin(__builtin_amdgcn_s_wait_asynccnt)
#define HAVE_ASYNC_LDS 1
#else
#define HAVE_ASYNC_LDS 0
#endif

#if HAVE_ASYNC_LDS
#define WAIT_ASYNC(n) __builtin_amdgcn_s_wait_asynccnt(n)
__device__ __forceinline__ void cp_async16(const __bf16* g, __bf16* l) {
  // generic->AS1: global VAs are identical; generic->AS3: low 32 bits are the
  // LDS offset (ISA aperture rule), so integer round-trips are value-correct.
  __builtin_amdgcn_global_load_async_to_lds_b128(
      (__attribute__((address_space(1))) v4i_vs*)(uintptr_t)g,
      (__attribute__((address_space(3))) v4i_vs*)(unsigned)(uintptr_t)l,
      0, 0);
}
#else
#define WAIT_ASYNC(n) ((void)0)
__device__ __forceinline__ void cp_async16(const __bf16* g, __bf16* l) {
  *(v8bf*)l = *(const v8bf*)g;
}
#endif

__device__ __forceinline__ v8f wmma_f32(v2f a, v2f b, v8f c) {
  return __builtin_amdgcn_wmma_f32_16x16x4_f32(false, a, false, b, (short)0, c,
                                               false, false);
}
__device__ __forceinline__ v8f wmma_bf16(v16bf a, v16bf b, v8f c) {
  return __builtin_amdgcn_wmma_f32_16x16x32_bf16(false, a, false, b, (short)0,
                                                 c, false, false);
}
// 16-element bf16 fragment: 8 bf16 at p (K = 8h..8h+7) + 8 bf16 at p+16
// (K = 16+8h..16+8h+7), per the 16-bit A/B operand layout.
__device__ __forceinline__ v16bf ldfrag(const __bf16* p) {
  v8bf c0 = *(const v8bf*)p;
  v8bf c1 = *(const v8bf*)(p + 16);
  return __builtin_shufflevector(c0, c1, 0, 1, 2, 3, 4, 5, 6, 7, 8, 9, 10, 11,
                                 12, 13, 14, 15);
}

// ---------------------------------------------------------------- gsum[i] = sum_j G[i][j]
__global__ __launch_bounds__(256) void gsum_kernel(const float* __restrict__ G,
                                                   float* __restrict__ gsum) {
  int row = blockIdx.x * 8 + (threadIdx.x >> 5);
  int lane = threadIdx.x & 31;
  const float* r = G + (size_t)row * IN_DIM;
  float acc = 0.f;
  for (int c = lane; c < IN_DIM; c += 32) acc += r[c];
  for (int off = 16; off > 0; off >>= 1) acc += __shfl_down(acc, off, 32);
  if (lane == 0) gsum[row] = acc;
}

// ---------------------------------------------------------------- split G into TRANSPOSED bf16 hi/lo planes
__global__ __launch_bounds__(256) void gconvT_kernel(const float* __restrict__ G,
                                                     __bf16* __restrict__ GhiT,
                                                     __bf16* __restrict__ GloT) {
  __shared__ __bf16 Hs[64 * LDT];
  __shared__ __bf16 Ls[64 * LDT];
  const int i0 = blockIdx.y * 64; // source rows (k of the GEMM)
  const int j0 = blockIdx.x * 64; // source cols (n of the GEMM)
  const int tid = threadIdx.x;
#pragma unroll
  for (int r = 0; r < 4; ++r) {
    int f = tid + (r << 8); // 1024 float4 chunks of the 64x64 tile
    int row = f >> 4, c4 = (f & 15) << 2;
    float4 v = *(const float4*)(G + (size_t)(i0 + row) * IN_DIM + j0 + c4);
    float x[4] = {v.x, v.y, v.z, v.w};
#pragma unroll
    for (int e = 0; e < 4; ++e) {
      __bf16 h = (__bf16)x[e];
      Hs[row * LDT + c4 + e] = h;
      Ls[row * LDT + c4 + e] = (__bf16)(x[e] - (float)h);
    }
  }
  __syncthreads();
#pragma unroll
  for (int r = 0; r < 2; ++r) {
    int f = tid + (r << 8); // 512 v8bf output chunks per plane
    int jr = f >> 3, i8 = (f & 7) << 3;
    v8bf h, l;
#pragma unroll
    for (int e = 0; e < 8; ++e) {
      h[e] = Hs[(i8 + e) * LDT + jr];
      l[e] = Ls[(i8 + e) * LDT + jr];
    }
    size_t o = (size_t)(j0 + jr) * IN_DIM + i0 + i8;
    *(v8bf*)(GhiT + o) = h;
    *(v8bf*)(GloT + o) = l;
  }
}

// ---------------------------------------------------------------- q[b][n][h] = sum_i s[n][b][i]*Qw[i][h]
__global__ __launch_bounds__(128) void q_kernel(const float* __restrict__ s,
                                                const float* __restrict__ Qw,
                                                float* __restrict__ q) {
  __shared__ float As[64 * LDA];
  __shared__ float Bs[16 * LDB];
  const int b = blockIdx.z;
  const int m0 = blockIdx.y * 64;
  const int n0 = blockIdx.x * 64;
  const int tid = threadIdx.x;
  const int lane = tid & 31;
  const int wid = tid >> 5;
  const int wm = (wid >> 1) << 5;
  const int wn = (wid & 1) << 5;
  const int lh = lane >> 4;
  const int l16 = lane & 15;

  v8f acc[2][2];
#pragma unroll
  for (int i = 0; i < 2; ++i)
#pragma unroll
    for (int j = 0; j < 2; ++j)
#pragma unroll
      for (int e = 0; e < 8; ++e) acc[i][j][e] = 0.f;

  for (int k0 = 0; k0 < IN_DIM; k0 += 16) {
#pragma unroll
    for (int r = 0; r < 2; ++r) { // A tile 64x16
      int f = tid + (r << 7);
      int row = f >> 2, c4 = (f & 3) << 2;
      float4 v = *(const float4*)(s + (size_t)(m0 + row) * SROW +
                                  (size_t)b * IN_DIM + k0 + c4);
      float* d = &As[row * LDA + c4];
      d[0] = v.x; d[1] = v.y; d[2] = v.z; d[3] = v.w;
    }
#pragma unroll
    for (int r = 0; r < 2; ++r) { // B tile 16x64 from Qw (ld = 64)
      int f = tid + (r << 7);
      int row = f >> 4, c4 = (f & 15) << 2;
      float4 v = *(const float4*)(Qw + (size_t)(k0 + row) * H_DIM + n0 + c4);
      float* d = &Bs[row * LDB + c4];
      d[0] = v.x; d[1] = v.y; d[2] = v.z; d[3] = v.w;
    }
    __syncthreads();
#pragma unroll
    for (int kk = 0; kk < 16; kk += 4) {
      v2f a[2], bb[2];
#pragma unroll
      for (int i = 0; i < 2; ++i) {
        const float* p = &As[(wm + (i << 4) + l16) * LDA + kk + (lh << 1)];
        a[i][0] = p[0]; a[i][1] = p[1];
      }
#pragma unroll
      for (int j = 0; j < 2; ++j) {
        int kb = kk + (lh << 1);
        bb[j][0] = Bs[kb * LDB + wn + (j << 4) + l16];
        bb[j][1] = Bs[(kb + 1) * LDB + wn + (j << 4) + l16];
      }
#pragma unroll
      for (int i = 0; i < 2; ++i)
#pragma unroll
        for (int j = 0; j < 2; ++j) acc[i][j] = wmma_f32(a[i], bb[j], acc[i][j]);
    }
    __syncthreads();
  }
#pragma unroll
  for (int i = 0; i < 2; ++i)
#pragma unroll
    for (int v = 0; v < 8; ++v) {
      int m = m0 + wm + (i << 4) + v + (lh << 3);
#pragma unroll
      for (int j = 0; j < 2; ++j) {
        int col = n0 + wn + (j << 4) + l16;
        q[(size_t)b * N_DIM * H_DIM + (size_t)m * H_DIM + col] = acc[i][j][v];
      }
    }
}

// ---------------------------------------------------------------- k[b][h][i] = sum_n Kw[n][h]*s[n][b][i]
__global__ __launch_bounds__(128) void k_kernel(const float* __restrict__ s,
                                                const float* __restrict__ Kw,
                                                float* __restrict__ kout) {
  __shared__ float As[64 * LDA];
  __shared__ float Bs[16 * LDB];
  const int b = blockIdx.z;
  const int n0 = blockIdx.x * 64;
  const int tid = threadIdx.x;
  const int lane = tid & 31;
  const int wid = tid >> 5;
  const int wm = (wid >> 1) << 5;
  const int wn = (wid & 1) << 5;
  const int lh = lane >> 4;
  const int l16 = lane & 15;

  v8f acc[2][2];
#pragma unroll
  for (int i = 0; i < 2; ++i)
#pragma unroll
    for (int j = 0; j < 2; ++j)
#pragma unroll
      for (int e = 0; e < 8; ++e) acc[i][j][e] = 0.f;

  for (int k0 = 0; k0 < N_DIM; k0 += 16) {
#pragma unroll
    for (int r = 0; r < 8; ++r) { // A = transpose of Kw[k0:k0+16][0:64]
      int e = tid + (r << 7);
      int kk = e >> 6, h = e & 63;
      As[h * LDA + kk] = Kw[(size_t)(k0 + kk) * H_DIM + h];
    }
#pragma unroll
    for (int r = 0; r < 2; ++r) { // B = s rows k0..k0+15 at batch b
      int f = tid + (r << 7);
      int row = f >> 4, c4 = (f & 15) << 2;
      float4 v = *(const float4*)(s + (size_t)(k0 + row) * SROW +
                                  (size_t)b * IN_DIM + n0 + c4);
      float* d = &Bs[row * LDB + c4];
      d[0] = v.x; d[1] = v.y; d[2] = v.z; d[3] = v.w;
    }
    __syncthreads();
#pragma unroll
    for (int kk = 0; kk < 16; kk += 4) {
      v2f a[2], bb[2];
#pragma unroll
      for (int i = 0; i < 2; ++i) {
        const float* p = &As[(wm + (i << 4) + l16) * LDA + kk + (lh << 1)];
        a[i][0] = p[0]; a[i][1] = p[1];
      }
#pragma unroll
      for (int j = 0; j < 2; ++j) {
        int kb = kk + (lh << 1);
        bb[j][0] = Bs[kb * LDB + wn + (j << 4) + l16];
        bb[j][1] = Bs[(kb + 1) * LDB + wn + (j << 4) + l16];
      }
#pragma unroll
      for (int i = 0; i < 2; ++i)
#pragma unroll
        for (int j = 0; j < 2; ++j) acc[i][j] = wmma_f32(a[i], bb[j], acc[i][j]);
    }
    __syncthreads();
  }
#pragma unroll
  for (int i = 0; i < 2; ++i)
#pragma unroll
    for (int v = 0; v < 8; ++v) {
      int m = wm + (i << 4) + v + (lh << 3); // h index
#pragma unroll
      for (int j = 0; j < 2; ++j) {
        int col = n0 + wn + (j << 4) + l16;
        kout[(size_t)b * H_DIM * IN_DIM + (size_t)m * IN_DIM + col] = acc[i][j][v];
      }
    }
}

// ---------------------------------------------------------------- att2 = (q.k/8 + 1e-9)^2 -> bf16 hi/lo planes
__global__ __launch_bounds__(128) void att_kernel(const float* __restrict__ q,
                                                  const float* __restrict__ k,
                                                  __bf16* __restrict__ ahi,
                                                  __bf16* __restrict__ alo) {
  __shared__ float As[64 * LDA];
  __shared__ float Bs[16 * LDB];
  const int b = blockIdx.z;
  const int m0 = blockIdx.y * 64;
  const int n0 = blockIdx.x * 64;
  const int tid = threadIdx.x;
  const int lane = tid & 31;
  const int wid = tid >> 5;
  const int wm = (wid >> 1) << 5;
  const int wn = (wid & 1) << 5;
  const int lh = lane >> 4;
  const int l16 = lane & 15;

  const float* qb = q + (size_t)b * N_DIM * H_DIM;
  const float* kb = k + (size_t)b * H_DIM * IN_DIM;

  v8f acc[2][2];
#pragma unroll
  for (int i = 0; i < 2; ++i)
#pragma unroll
    for (int j = 0; j < 2; ++j)
#pragma unroll
      for (int e = 0; e < 8; ++e) acc[i][j][e] = 0.f;

  for (int k0 = 0; k0 < H_DIM; k0 += 16) {
#pragma unroll
    for (int r = 0; r < 2; ++r) { // A 64x16 from q (ld 64)
      int f = tid + (r << 7);
      int row = f >> 2, c4 = (f & 3) << 2;
      float4 v = *(const float4*)(qb + (size_t)(m0 + row) * H_DIM + k0 + c4);
      float* d = &As[row * LDA + c4];
      d[0] = v.x; d[1] = v.y; d[2] = v.z; d[3] = v.w;
    }
#pragma unroll
    for (int r = 0; r < 2; ++r) { // B 16x64 from k (ld 1024)
      int f = tid + (r << 7);
      int row = f >> 4, c4 = (f & 15) << 2;
      float4 v = *(const float4*)(kb + (size_t)(k0 + row) * IN_DIM + n0 + c4);
      float* d = &Bs[row * LDB + c4];
      d[0] = v.x; d[1] = v.y; d[2] = v.z; d[3] = v.w;
    }
    __syncthreads();
#pragma unroll
    for (int kk = 0; kk < 16; kk += 4) {
      v2f a[2], bb[2];
#pragma unroll
      for (int i = 0; i < 2; ++i) {
        const float* p = &As[(wm + (i << 4) + l16) * LDA + kk + (lh << 1)];
        a[i][0] = p[0]; a[i][1] = p[1];
      }
#pragma unroll
      for (int j = 0; j < 2; ++j) {
        int kb2 = kk + (lh << 1);
        bb[j][0] = Bs[kb2 * LDB + wn + (j << 4) + l16];
        bb[j][1] = Bs[(kb2 + 1) * LDB + wn + (j << 4) + l16];
      }
#pragma unroll
      for (int i = 0; i < 2; ++i)
#pragma unroll
        for (int j = 0; j < 2; ++j) acc[i][j] = wmma_f32(a[i], bb[j], acc[i][j]);
    }
    __syncthreads();
  }
  const float inv_sqrt_h = 0.125f;
#pragma unroll
  for (int i = 0; i < 2; ++i)
#pragma unroll
    for (int v = 0; v < 8; ++v) {
      int m = m0 + wm + (i << 4) + v + (lh << 3);
#pragma unroll
      for (int j = 0; j < 2; ++j) {
        int col = n0 + wn + (j << 4) + l16;
        float t = acc[i][j][v] * inv_sqrt_h + 1e-9f;
        float t2 = t * t;
        __bf16 h = (__bf16)t2;
        size_t o = (size_t)b * N_DIM * IN_DIM + (size_t)m * IN_DIM + col;
        ahi[o] = h;
        alo[o] = (__bf16)(t2 - (float)h);
      }
    }
}

// ---------------------------------------------------------------- st[b][n] = att2[b][n][:] . gsum + 0.001
__global__ __launch_bounds__(256) void st_kernel(const __bf16* __restrict__ ahi,
                                                 const __bf16* __restrict__ alo,
                                                 const float* __restrict__ gsum,
                                                 float* __restrict__ st) {
  int idx = blockIdx.x * 8 + (threadIdx.x >> 5); // b*1024 + n
  int lane = threadIdx.x & 31;
  const __bf16* rh = ahi + (size_t)idx * IN_DIM;
  const __bf16* rl = alo + (size_t)idx * IN_DIM;
  float acc = 0.f;
  for (int c = lane; c < IN_DIM; c += 32)
    acc += ((float)rh[c] + (float)rl[c]) * gsum[c];
  for (int off = 16; off > 0; off >>= 1) acc += __shfl_down(acc, off, 32);
  if (lane == 0) st[idx] = acc + 0.001f;
}

// ---------------------------------------------------------------- out = (att2 @ G) / st, bf16x3 split GEMM
// Async double-buffered: all four plane tiles are verbatim row copies
// (A planes row-major [m][k]; B planes from pre-transposed GT [n][k]).
__global__ __launch_bounds__(128) void attg_kernel(
    const __bf16* __restrict__ ahi, const __bf16* __restrict__ alo,
    const __bf16* __restrict__ GhiT, const __bf16* __restrict__ GloT,
    const float* __restrict__ st, float* __restrict__ out) {
  __shared__ __bf16 sm[2][4][64 * LDK]; // [buf][plane: Ah Al Bh Bl][row*LDK+k]
  const int b = blockIdx.z;
  const int m0 = blockIdx.y * 64;
  const int n0 = blockIdx.x * 64;
  const int tid = threadIdx.x;
  const int lane = tid & 31;
  const int wid = tid >> 5;
  const int wm = (wid >> 1) << 5;
  const int wn = (wid & 1) << 5;
  const int lh = lane >> 4;
  const int l16 = lane & 15;

  const __bf16* A0 = ahi + (size_t)b * N_DIM * IN_DIM + (size_t)m0 * IN_DIM;
  const __bf16* A1 = alo + (size_t)b * N_DIM * IN_DIM + (size_t)m0 * IN_DIM;
  const __bf16* B0 = GhiT + (size_t)n0 * IN_DIM;
  const __bf16* B1 = GloT + (size_t)n0 * IN_DIM;

  // per-thread chunk coordinates (2 chunks x 4 planes = 8 async ops / stage)
  const int row0 = tid >> 2, c80 = (tid & 3) << 3;
  const int row1 = (tid + 128) >> 2, c81 = ((tid + 128) & 3) << 3;

  v8f acc[2][2];
#pragma unroll
  for (int i = 0; i < 2; ++i)
#pragma unroll
    for (int j = 0; j < 2; ++j)
#pragma unroll
      for (int e = 0; e < 8; ++e) acc[i][j][e] = 0.f;

#define STAGE_LOAD(P, K0)                                                   \
  do {                                                                      \
    size_t g0 = (size_t)row0 * IN_DIM + (K0) + c80;                         \
    size_t g1 = (size_t)row1 * IN_DIM + (K0) + c81;                         \
    int l0 = row0 * LDK + c80, l1 = row1 * LDK + c81;                       \
    cp_async16(A0 + g0, &sm[P][0][l0]);                                     \
    cp_async16(A0 + g1, &sm[P][0][l1]);                                     \
    cp_async16(A1 + g0, &sm[P][1][l0]);                                     \
    cp_async16(A1 + g1, &sm[P][1][l1]);                                     \
    cp_async16(B0 + g0, &sm[P][2][l0]);                                     \
    cp_async16(B0 + g1, &sm[P][2][l1]);                                     \
    cp_async16(B1 + g0, &sm[P][3][l0]);                                     \
    cp_async16(B1 + g1, &sm[P][3][l1]);                                     \
  } while (0)

  STAGE_LOAD(0, 0);
  int p = 0;
  for (int k0 = 0; k0 < IN_DIM; k0 += 32, p ^= 1) {
    if (k0 + 32 < IN_DIM) {
      STAGE_LOAD(p ^ 1, k0 + 32);
      WAIT_ASYNC(8); // oldest 8 (current stage) complete; async is in-order
    } else {
      WAIT_ASYNC(0);
    }
    __syncthreads();
    v16bf fah[2], fal[2], fbh[2], fbl[2];
#pragma unroll
    for (int i = 0; i < 2; ++i) {
      int row = wm + (i << 4) + l16;
      fah[i] = ldfrag(&sm[p][0][row * LDK + (lh << 3)]);
      fal[i] = ldfrag(&sm[p][1][row * LDK + (lh << 3)]);
    }
#pragma unroll
    for (int j = 0; j < 2; ++j) {
      int row = wn + (j << 4) + l16;
      fbh[j] = ldfrag(&sm[p][2][row * LDK + (lh << 3)]);
      fbl[j] = ldfrag(&sm[p][3][row * LDK + (lh << 3)]);
    }
#pragma unroll
    for (int i = 0; i < 2; ++i)
#pragma unroll
      for (int j = 0; j < 2; ++j) {
        acc[i][j] = wmma_bf16(fah[i], fbh[j], acc[i][j]);
        acc[i][j] = wmma_bf16(fah[i], fbl[j], acc[i][j]);
        acc[i][j] = wmma_bf16(fal[i], fbh[j], acc[i][j]);
      }
    __syncthreads();
  }
#undef STAGE_LOAD
#pragma unroll
  for (int i = 0; i < 2; ++i)
#pragma unroll
    for (int v = 0; v < 8; ++v) {
      int m = m0 + wm + (i << 4) + v + (lh << 3);
      float inv = 1.0f / st[(size_t)b * N_DIM + m];
#pragma unroll
      for (int j = 0; j < 2; ++j) {
        int col = n0 + wn + (j << 4) + l16;
        out[(size_t)m * (B_DIM * IN_DIM) + (size_t)b * IN_DIM + col] =
            acc[i][j][v] * inv;
      }
    }
}

extern "C" void kernel_launch(void* const* d_in, const int* in_sizes, int n_in,
                              void* d_out, int out_size, void* d_ws,
                              size_t ws_size, hipStream_t stream) {
  (void)in_sizes; (void)n_in; (void)out_size; (void)ws_size;
  const float* s = (const float*)d_in[0];
  const float* G = (const float*)d_in[1];
  const float* Qw = (const float*)d_in[2];
  const float* Kw = (const float*)d_in[3];
  float* out = (float*)d_out;

  float* ws = (float*)d_ws;
  float* q = ws;                                // 16*1024*64 f32
  float* k = q + (size_t)B_DIM * N_DIM * H_DIM; // 16*64*1024 f32
  __bf16* ahi = (__bf16*)(k + (size_t)B_DIM * H_DIM * IN_DIM); // 16M bf16
  __bf16* alo = ahi + (size_t)B_DIM * N_DIM * IN_DIM;          // 16M bf16
  __bf16* GhiT = alo + (size_t)B_DIM * N_DIM * IN_DIM;         // 1M bf16
  __bf16* GloT = GhiT + (size_t)IN_DIM * IN_DIM;               // 1M bf16
  float* gsum = (float*)(GloT + (size_t)IN_DIM * IN_DIM);      // 1024 f32
  float* st = gsum + IN_DIM;                                   // 16*1024 f32

  gsum_kernel<<<IN_DIM / 8, 256, 0, stream>>>(G, gsum);
  gconvT_kernel<<<dim3(IN_DIM / 64, IN_DIM / 64), 256, 0, stream>>>(G, GhiT, GloT);
  q_kernel<<<dim3(1, N_DIM / 64, B_DIM), 128, 0, stream>>>(s, Qw, q);
  k_kernel<<<dim3(IN_DIM / 64, 1, B_DIM), 128, 0, stream>>>(s, Kw, k);
  att_kernel<<<dim3(IN_DIM / 64, N_DIM / 64, B_DIM), 128, 0, stream>>>(q, k, ahi, alo);
  st_kernel<<<(B_DIM * N_DIM) / 8, 256, 0, stream>>>(ahi, alo, gsum, st);
  attg_kernel<<<dim3(IN_DIM / 64, N_DIM / 64, B_DIM), 128, 0, stream>>>(
      ahi, alo, GhiT, GloT, st, out);
}
